// ResidualNet_1228360646806
// MI455X (gfx1250) — matrix operand order repaired and checked by reading, not compiled, not run on hardware
//
#include <hip/hip_runtime.h>

typedef __bf16 bf16;
typedef __attribute__((ext_vector_type(16))) __bf16 v16bf;
typedef __attribute__((ext_vector_type(8)))  float  v8f;

#define T_DIM 128
#define B_DIM 64
#define V_DIM 10000
#define E_DIM 512
#define H_DIM 2048
#define L_DIM 3
#define M_DIM (T_DIM * B_DIM) /* 8192 */
#define EPS_F 1e-5f

#define BM 128
#define BN 128
#define BK 32
#define KPAD 8
#define GEMM_BIAS 1
#define GEMM_RES  2

// ---------------- CDNA5 async global->LDS helpers ----------------

__device__ __forceinline__ void async_ld_b128(unsigned lds_off, const void* gptr) {
  asm volatile("global_load_async_to_lds_b128 %0, %1, off"
               :: "v"(lds_off), "v"((unsigned long long)(uintptr_t)gptr)
               : "memory");
}
__device__ __forceinline__ void wait_async0() {
  asm volatile("s_wait_asynccnt 0x0" ::: "memory");
}
__device__ __forceinline__ unsigned lds_off_of(const void* p) {
  // generic pointer to LDS: addr[31:0] is the workgroup-relative LDS offset
  return (unsigned)(uintptr_t)p;
}

// ---------------- elementwise / gather ----------------

__global__ void embed_kernel(const int* __restrict__ tokens, const float* __restrict__ emb,
                             bf16* __restrict__ out) {
  int i = blockIdx.x * blockDim.x + threadIdx.x;  // m*E + e
  if (i >= M_DIM * E_DIM) return;
  int m = i >> 9;              // / E_DIM (512)
  int e = i & (E_DIM - 1);
  out[i] = (bf16)emb[(size_t)tokens[m] * E_DIM + e];
}

// fused fp32->bf16 + transpose:  in [K][N] (f32, row-major)  ->  out [N][K] (bf16, row-major)
// grid: (ceil(N/32), K/32, slices); block 256 = 32x8
__global__ void wtrans_kernel(const float* __restrict__ in, bf16* __restrict__ out,
                              int K, int N, size_t in_slice, size_t out_slice) {
  const float* src = in  + (size_t)blockIdx.z * in_slice;
  bf16*        dst = out + (size_t)blockIdx.z * out_slice;
  __shared__ float tile[32][33];
  const int n0 = blockIdx.x * 32, k0 = blockIdx.y * 32;
  const int tx = threadIdx.x & 31, ty = threadIdx.x >> 5;
#pragma unroll
  for (int i = ty; i < 32; i += 8) {
    int k = k0 + i, n = n0 + tx;
    if (n < N) tile[i][tx] = src[(size_t)k * N + n];   // K is always a multiple of 32
  }
  __syncthreads();
#pragma unroll
  for (int i = ty; i < 32; i += 8) {
    int n = n0 + i, k = k0 + tx;
    if (n < N) dst[(size_t)n * K + k] = (bf16)tile[tx][i];
  }
}

// per-(t,c) batchnorm statistics over the batch axis (population variance)
__global__ void bn_stats_kernel(const float* __restrict__ X, float* __restrict__ meanp,
                                float* __restrict__ rstdp, int C) {
  int i = blockIdx.x * blockDim.x + threadIdx.x;  // t*C + c
  if (i >= T_DIM * C) return;
  int t = i / C;
  int c = i - t * C;
  const float* p = X + (size_t)t * B_DIM * C + c;
  float s = 0.f, s2 = 0.f;
  for (int b = 0; b < B_DIM; ++b) {
    float v = p[(size_t)b * C];
    s += v; s2 += v * v;
  }
  float mu  = s * (1.f / B_DIM);
  float var = s2 * (1.f / B_DIM) - mu * mu;
  meanp[i] = mu;
  rstdp[i] = rsqrtf(var + EPS_F);
}

// BN (precomputed stats) + IndRNN scan over T; optional bf16 / f32 outputs
__global__ void indrnn_scan_kernel(const float* __restrict__ X,
                                   const float* __restrict__ meanp, const float* __restrict__ rstdp,
                                   const float* __restrict__ g, const float* __restrict__ beta,
                                   const float* __restrict__ u,
                                   bf16* __restrict__ out_bf, float* __restrict__ out_f, int C) {
  int i = blockIdx.x * blockDim.x + threadIdx.x;  // b*C + c
  if (i >= B_DIM * C) return;
  int b = i / C;
  int c = i - b * C;
  float gg = g[c], bb = beta[c], uu = u[c];
  float h = 0.f;
  for (int t = 0; t < T_DIM; ++t) {
    size_t off = ((size_t)t * B_DIM + b) * C + c;
    int si = t * C + c;
    float v = (X[off] - meanp[si]) * rstdp[si] * gg + bb;
    h = v + uu * h;
    h = h > 0.f ? h : 0.f;
    if (out_bf) out_bf[off] = (bf16)h;
    if (out_f)  out_f[off]  = h;
  }
}

// plain BN apply (extra_bn), emits bf16 for the decoder GEMM
__global__ void bn_apply_kernel(const float* __restrict__ X,
                                const float* __restrict__ meanp, const float* __restrict__ rstdp,
                                const float* __restrict__ g, const float* __restrict__ beta,
                                bf16* __restrict__ out, int C) {
  int i = blockIdx.x * blockDim.x + threadIdx.x;  // m*C + c
  if (i >= M_DIM * C) return;
  int m = i / C;
  int c = i - m * C;
  int t = m / B_DIM;
  int si = t * C + c;
  out[i] = (bf16)((X[i] - meanp[si]) * rstdp[si] * g[c] + beta[c]);
}

// ---------------- WMMA GEMM ----------------
// C[M,N] = A[M,K](bf16,row) @ Wt[N,K](bf16,row, pre-transposed) (+bias)(+residual)
// 256 threads = 8 waves (4 m x 2 n); tile 128x128, K-step 32; each wave 32x64 = 2x4 WMMA accs.
// Tiles staged with global_load_async_to_lds_b128, double-buffered (ASYNCcnt).

__global__ __launch_bounds__(256)
void gemm_bf16_wmma_kernel(const bf16* __restrict__ A, const bf16* __restrict__ Wt,
                           const float* __restrict__ bias, const float* __restrict__ Cin,
                           float* __restrict__ Cout, int N, int K, int flags) {
  __shared__ __align__(16) bf16 As[2][BM][BK + KPAD];
  __shared__ __align__(16) bf16 Bt[2][BN][BK + KPAD];

  const int tid  = threadIdx.x;
  const int lane = tid & 31;
  const int wid  = tid >> 5;
  const int wm   = wid & 3;        // wave row 0..3
  const int wn   = wid >> 2;       // wave col 0..1
  const int hf   = lane >> 4;      // lane half
  const int l16  = lane & 15;

  const int bm0 = blockIdx.y * BM;
  const int bn0 = blockIdx.x * BN;

  // staging map: each thread owns 16 consecutive bf16 of one tile row
  const int s_row = tid >> 1;            // 0..127
  const int s_col = (tid & 1) << 4;      // 0 or 16

  const bf16* gA = A  + (size_t)(bm0 + s_row) * K + s_col;
  const bf16* gB = Wt + (size_t)(bn0 + s_row) * K + s_col;
  const bool  bvalid = (bn0 + s_row) < N;    // decoder tail rows

  v8f acc[2][4];
#pragma unroll
  for (int a = 0; a < 2; ++a)
#pragma unroll
    for (int f = 0; f < 4; ++f)
#pragma unroll
      for (int i = 0; i < 8; ++i) acc[a][f][i] = 0.f;

  // ---- async stage of one K-tile into buffer `bf` ----
  auto stage = [&](int bsel, int k0) {
    unsigned la = lds_off_of(&As[bsel][s_row][s_col]);
    async_ld_b128(la,      gA + k0);
    async_ld_b128(la + 16, gA + k0 + 8);
    if (bvalid) {
      unsigned lb = lds_off_of(&Bt[bsel][s_row][s_col]);
      async_ld_b128(lb,      gB + k0);
      async_ld_b128(lb + 16, gB + k0 + 8);
    }
  };

  stage(0, 0);
  wait_async0();
  __syncthreads();

  int buf = 0;
  for (int k0 = 0; k0 < K; k0 += BK) {
    if (k0 + BK < K) stage(buf ^ 1, k0 + BK);   // overlap DMA with compute

    // ---- fragments (all contiguous 16B LDS reads) ----
    v16bf af[2];
    v16bf bfv[4];
#pragma unroll
    for (int a = 0; a < 2; ++a) {
      const int row = wm * 32 + a * 16 + l16;
#pragma unroll
      for (int p = 0; p < 8; ++p) {
        // A 16x32, this code's contiguous k order (B matches): lanes0-15 hold K 0..15, lanes16-31 hold K 16..31
        const int kk = 2 * p + 16 * hf;
        af[a][2 * p]     = As[buf][row][kk];
        af[a][2 * p + 1] = As[buf][row][kk + 1];
      }
    }
#pragma unroll
    for (int f = 0; f < 4; ++f) {
      const int col = wn * 64 + f * 16 + l16;
#pragma unroll
      for (int j = 0; j < 16; ++j) {
        // B 32x16, this code's contiguous k order (A matches): lanes0-15 hold K 0..15, lanes16-31 hold K 16..31
        bfv[f][j] = Bt[buf][col][j + 16 * hf];
      }
    }

#pragma unroll
    for (int a = 0; a < 2; ++a)
#pragma unroll
      for (int f = 0; f < 4; ++f)
        acc[a][f] = __builtin_amdgcn_wmma_f32_16x16x32_bf16(
            false, af[a], false, bfv[f], (short)0, acc[a][f], false, false);

    wait_async0();
    __syncthreads();
    buf ^= 1;
  }

  // ---- epilogue: D VGPR i -> row (i + 8*half), col l16 ----
#pragma unroll
  for (int a = 0; a < 2; ++a) {
#pragma unroll
    for (int f = 0; f < 4; ++f) {
      const int n = bn0 + wn * 64 + f * 16 + l16;
      if (n < N) {
#pragma unroll
        for (int i = 0; i < 8; ++i) {
          const int m = bm0 + wm * 32 + a * 16 + 8 * hf + i;
          float v = acc[a][f][i];
          if (flags & GEMM_BIAS) v += bias[n];
          const size_t off = (size_t)m * N + n;
          if (flags & GEMM_RES) v += Cin[off];
          Cout[off] = v;
        }
      }
    }
  }
}

// ---------------- host orchestration ----------------

extern "C" void kernel_launch(void* const* d_in, const int* in_sizes, int n_in,
                              void* d_out, int out_size, void* d_ws, size_t ws_size,
                              hipStream_t stream) {
  (void)in_sizes; (void)n_in; (void)out_size; (void)ws_size;

  const int*   tokens = (const int*)  d_in[0];
  const float* emb    = (const float*)d_in[1];
  const float* fc0_w  = (const float*)d_in[2];
  const float* fc0_b  = (const float*)d_in[3];
  const float* bn1_g  = (const float*)d_in[4];
  const float* bn1_b  = (const float*)d_in[5];
  const float* u1     = (const float*)d_in[6];
  const float* fc1_w  = (const float*)d_in[7];
  const float* fc1_b  = (const float*)d_in[8];
  const float* bn2_g  = (const float*)d_in[9];
  const float* bn2_b  = (const float*)d_in[10];
  const float* u2     = (const float*)d_in[11];
  const float* fc2_w  = (const float*)d_in[12];
  const float* fc2_b  = (const float*)d_in[13];
  const float* rf_g   = (const float*)d_in[14];
  const float* rf_b   = (const float*)d_in[15];
  const float* rf_u   = (const float*)d_in[16];
  const float* last_w = (const float*)d_in[17];
  const float* last_b = (const float*)d_in[18];
  const float* li_g   = (const float*)d_in[19];
  const float* li_b   = (const float*)d_in[20];
  const float* li_u   = (const float*)d_in[21];
  const float* ebn_g  = (const float*)d_in[22];
  const float* ebn_b  = (const float*)d_in[23];
  const float* dec_w  = (const float*)d_in[24];
  const float* dec_b  = (const float*)d_in[25];

  // ---- workspace carve (deterministic layout, ~243 MB) ----
  char* base = (char*)d_ws;
  size_t off = 0;
  auto carve = [&](size_t bytes) -> void* {
    void* p = base + off;
    off += (bytes + 255) & ~(size_t)255;
    return p;
  };
  // all weights stored transposed: Wt[N][K] bf16
  bf16*  wfc0  = (bf16*) carve((size_t)H_DIM * E_DIM * 2);           // [H][E]
  bf16*  wfc1  = (bf16*) carve((size_t)L_DIM * H_DIM * H_DIM * 2);   // [H][H] x L
  bf16*  wfc2  = (bf16*) carve((size_t)L_DIM * H_DIM * H_DIM * 2);
  bf16*  wlast = (bf16*) carve((size_t)E_DIM * H_DIM * 2);           // [E][H]
  bf16*  wdec  = (bf16*) carve((size_t)V_DIM * E_DIM * 2);           // [V][E]
  float* X     = (float*)carve((size_t)M_DIM * H_DIM * 4);
  float* O2    = (float*)carve((size_t)M_DIM * H_DIM * 4);   // also reused as Y [M,E]
  float* Z     = (float*)carve((size_t)M_DIM * E_DIM * 4);
  bf16*  Abf   = (bf16*) carve((size_t)M_DIM * H_DIM * 2);   // bf16 GEMM A operand
  float* meanb = (float*)carve((size_t)T_DIM * H_DIM * 4);
  float* rstdb = (float*)carve((size_t)T_DIM * H_DIM * 4);

  auto cdiv = [](int a, int b) { return (a + b - 1) / b; };

  auto gemm = [&](const bf16* Ain, const bf16* Win, const float* bias,
                  const float* Cin, float* Cout, int N, int K, int flags) {
    dim3 grid(cdiv(N, BN), M_DIM / BM);
    gemm_bf16_wmma_kernel<<<grid, 256, 0, stream>>>(Ain, Win, bias, Cin, Cout, N, K, flags);
  };
  auto wtrans = [&](const float* w, bf16* wt, int K, int N, int slices, size_t islice, size_t oslice) {
    dim3 grid(cdiv(N, 32), K / 32, slices);
    wtrans_kernel<<<grid, 256, 0, stream>>>(w, wt, K, N, islice, oslice);
  };

  // ---- 1. weights: fp32 -> bf16, transposed to [N][K] ----
  wtrans(fc0_w,  wfc0,  E_DIM, H_DIM, 1,     0, 0);
  wtrans(fc1_w,  wfc1,  H_DIM, H_DIM, L_DIM, (size_t)H_DIM * H_DIM, (size_t)H_DIM * H_DIM);
  wtrans(fc2_w,  wfc2,  H_DIM, H_DIM, L_DIM, (size_t)H_DIM * H_DIM, (size_t)H_DIM * H_DIM);
  wtrans(last_w, wlast, H_DIM, E_DIM, 1,     0, 0);
  wtrans(dec_w,  wdec,  E_DIM, V_DIM, 1,     0, 0);

  // ---- 2. embedding gather -> bf16, then fc0 ----
  embed_kernel<<<cdiv(M_DIM * E_DIM, 256), 256, 0, stream>>>(tokens, emb, Abf);
  gemm(Abf, wfc0, fc0_b, nullptr, X, H_DIM, E_DIM, GEMM_BIAS);

  // ---- 3. residual blocks ----
  for (int i = 0; i < L_DIM; ++i) {
    const size_t wOff = (size_t)i * H_DIM * H_DIM;
    const size_t pOff = (size_t)i * H_DIM;
    bn_stats_kernel<<<cdiv(T_DIM * H_DIM, 256), 256, 0, stream>>>(X, meanb, rstdb, H_DIM);
    indrnn_scan_kernel<<<cdiv(B_DIM * H_DIM, 256), 256, 0, stream>>>(
        X, meanb, rstdb, bn1_g + pOff, bn1_b + pOff, u1 + pOff, Abf, (float*)nullptr, H_DIM);
    gemm(Abf, wfc1 + wOff, fc1_b + pOff, nullptr, O2, H_DIM, H_DIM, GEMM_BIAS);

    bn_stats_kernel<<<cdiv(T_DIM * H_DIM, 256), 256, 0, stream>>>(O2, meanb, rstdb, H_DIM);
    indrnn_scan_kernel<<<cdiv(B_DIM * H_DIM, 256), 256, 0, stream>>>(
        O2, meanb, rstdb, bn2_g + pOff, bn2_b + pOff, u2 + pOff, Abf, (float*)nullptr, H_DIM);
    gemm(Abf, wfc2 + wOff, fc2_b + pOff, X, X, H_DIM, H_DIM, GEMM_BIAS | GEMM_RES);
  }

  // ---- 4. rf block + last_fc ----
  bn_stats_kernel<<<cdiv(T_DIM * H_DIM, 256), 256, 0, stream>>>(X, meanb, rstdb, H_DIM);
  indrnn_scan_kernel<<<cdiv(B_DIM * H_DIM, 256), 256, 0, stream>>>(
      X, meanb, rstdb, rf_g, rf_b, rf_u, Abf, (float*)nullptr, H_DIM);
  gemm(Abf, wlast, last_b, nullptr, O2 /*Y [M,E]*/, E_DIM, H_DIM, GEMM_BIAS);

  // ---- 5. li indrnn_bn (fp32 out), extra bn -> bf16 ----
  bn_stats_kernel<<<cdiv(T_DIM * E_DIM, 256), 256, 0, stream>>>(O2, meanb, rstdb, E_DIM);
  indrnn_scan_kernel<<<cdiv(B_DIM * E_DIM, 256), 256, 0, stream>>>(
      O2, meanb, rstdb, li_g, li_b, li_u, (bf16*)nullptr, Z, E_DIM);
  bn_stats_kernel<<<cdiv(T_DIM * E_DIM, 256), 256, 0, stream>>>(Z, meanb, rstdb, E_DIM);
  bn_apply_kernel<<<cdiv(M_DIM * E_DIM, 256), 256, 0, stream>>>(Z, meanb, rstdb, ebn_g, ebn_b, Abf, E_DIM);

  // ---- 6. decoder -> d_out (fp32 logits) ----
  gemm(Abf, wdec, dec_b, nullptr, (float*)d_out, V_DIM, E_DIM, GEMM_BIAS);
}
